// MultiScalePhysicsTransformer_58007828299844
// MI455X (gfx1250) — matrix operand order repaired and checked by reading, not compiled, not run on hardware
//
#include <hip/hip_runtime.h>

// ---------------------------------------------------------------------------
// MultiScalePhysicsTransformer for MI455X (gfx1250, wave32, WMMA).
// GEMM-shaped work: v_wmma_f32_16x16x32_f16 fed from LDS tiles staged with
// global_load_async_to_lds_b128 (ASYNCcnt). Elementwise/scan/norm: f32 VALU.
// ---------------------------------------------------------------------------

typedef __attribute__((ext_vector_type(16))) _Float16 v16h;
typedef __attribute__((ext_vector_type(8)))  float    v8f;

static constexpr int BSZ = 16;
static constexpr int SEQ = 1024;
static constexpr int BL  = BSZ * SEQ;
static constexpr int DM  = 256;    // D_MODEL
static constexpr int DI  = 512;    // D_INNER
static constexpr int DS  = 32;     // D_STATE
static constexpr int DTR = 16;     // DT_RANK

__device__ __forceinline__ unsigned lds_off32(const void* p)
{
    // addrspace(3) generic address: low 32 bits are the LDS byte offset
    return (unsigned)(size_t)p;
}

__device__ __forceinline__ void async_ld_b128(unsigned lds, const void* gaddr)
{
    asm volatile("global_load_async_to_lds_b128 %0, %1, off"
                 :: "v"(lds), "v"(gaddr) : "memory");
}

__device__ __forceinline__ void wait_async0()
{
    asm volatile("s_wait_asynccnt 0" ::: "memory");
}

// ===========================================================================
// WMMA GEMM:  C[M,N] = act( A[M,K](f16, row stride lda) * W[N,K](f16)^T + bias )
// Block = 128 threads (4 waves) -> 16x64 C tile; K stepped by 32.
// A (1KB) and W (4KB) tiles staged in LDS via async-to-LDS copies.
// ===========================================================================
template <int ACT>  // 0 = none, 1 = relu
__global__ __launch_bounds__(128)
void gemm_wmma(const _Float16* __restrict__ A, int lda,
               const _Float16* __restrict__ W,
               const float* __restrict__ bias,
               float* __restrict__ C,
               int M, int N, int K)
{
    __shared__ alignas(16) _Float16 sA[16 * 32];
    __shared__ alignas(16) _Float16 sW[64 * 32];

    const int lane = threadIdx.x;            // 0..31
    const int wv   = threadIdx.y;            // 0..3
    const int tid  = wv * 32 + lane;
    const int m0   = blockIdx.x * 16;
    const int nblk = blockIdx.y * 64;
    const int n0   = nblk + wv * 16;

    const int row = lane & 15;
    const int hi  = lane >> 4;               // 0 | 1

    const int ar = tid >> 2, ac = tid & 3;   // A-tile staging: row, 16B chunk

    v8f acc = {};
    for (int kb = 0; kb < K; kb += 32) {
        // ---- stage A tile (16 rows x 32 halves = 64 x 16B chunks) --------
        if (tid < 64) {
            const _Float16* gp = A + (size_t)(m0 + ar) * lda + kb + ac * 8;
            async_ld_b128(lds_off32(&sA[ar * 32 + ac * 8]), gp);
        }
        // ---- stage W tile (64 rows x 32 halves = 256 chunks, 2/thread) ---
#pragma unroll
        for (int i = 0; i < 2; ++i) {
            int id = tid + i * 128;
            int wr = id >> 2, wc = id & 3;
            const _Float16* gp = W + (size_t)(nblk + wr) * K + kb + wc * 8;
            async_ld_b128(lds_off32(&sW[wr * 32 + wc * 8]), gp);
        }
        __builtin_prefetch(A + (size_t)(m0 + ar) * lda + kb + 32, 0, 1);
        wait_async0();
        __syncthreads();

        // ---- fragments from LDS ------------------------------------------
        v16h a, b;
#pragma unroll
        for (int e = 0; e < 16; ++e) {
            // A frag (16x32): lanes0-15 K=0..15 ; lanes16-31 K=16..31
            int k = hi * 16 + e;
            a[e] = sA[row * 32 + k];
        }
#pragma unroll
        for (int e = 0; e < 16; ++e) {
            // B frag (32x16): lanes0-15 K=0..15 ; lanes16-31 K=16..31
            int k = hi * 16 + e;
            b[e] = (kb + k < K) ? sW[(wv * 16 + row) * 32 + k] : (_Float16)0.f;
        }
        acc = __builtin_amdgcn_wmma_f32_16x16x32_f16(false, a, false, b,
                                                     (short)0, acc, false, false);
        __syncthreads();
    }

    if (n0 < N) {   // wave-uniform store guard (N=80 partial block)
#pragma unroll
        for (int r = 0; r < 8; ++r) {
            int m = m0 + r + hi * 8;
            int n = n0 + row;
            float v = acc[r];
            if (bias) v += bias[n];
            if (ACT == 1) v = fmaxf(v, 0.f);
            C[(size_t)m * N + n] = v;
        }
    }
}

// ===========================================================================
// Flash attention, one wave per (b, head, 16-query tile).
// S = (Q/sqrt(d)) K^T via WMMA, online softmax with width-16 shuffles,
// P re-layout via 2KB LDS, O += P V via WMMA.  HEAD_DIM = 32.
// ===========================================================================
__global__ __launch_bounds__(32)
void attention_flash(const float* __restrict__ qkv,  // [B*L, 768]
                     float* __restrict__ o)          // [B*L, 256]
{
    __shared__ float P[16 * 32];
    const int lane = threadIdx.x;
    const int row  = lane & 15;
    const int hi   = lane >> 4;
    const int lt   = blockIdx.x;      // query tile
    const int h    = blockIdx.y;      // head
    const int b    = blockIdx.z;      // batch
    const float scale = 0.17677669529663687f;   // 1/sqrt(32)

    v16h qa;
    {
        const float* qp = qkv + ((size_t)(b * SEQ) + lt * 16 + row) * 768 + h * 32;
#pragma unroll
        for (int e = 0; e < 16; ++e) {
            int k = hi * 16 + e;
            qa[e] = (_Float16)(qp[k] * scale);
        }
    }

    v8f o0 = {}, o1 = {};
    float Mx[8], Sm[8];
#pragma unroll
    for (int r = 0; r < 8; ++r) { Mx[r] = -1e30f; Sm[r] = 0.f; }

    for (int ms = 0; ms < SEQ; ms += 32) {
        // ---- scores for keys ms..ms+31 -------------------------------------
        v16h kb0, kb1;
        const float* kp = qkv + ((size_t)(b * SEQ) + ms) * 768 + 256 + h * 32;
#pragma unroll
        for (int e = 0; e < 16; ++e) {
            int kk = hi * 16 + e;                       // head-dim index
            kb0[e] = (_Float16)kp[(size_t)row * 768 + kk];
            kb1[e] = (_Float16)kp[(size_t)(16 + row) * 768 + kk];
        }
        v8f s0 = {}, s1 = {};
        s0 = __builtin_amdgcn_wmma_f32_16x16x32_f16(false, qa, false, kb0, (short)0, s0, false, false);
        s1 = __builtin_amdgcn_wmma_f32_16x16x32_f16(false, qa, false, kb1, (short)0, s1, false, false);

        // ---- online softmax (row stats across the 16 lanes of each row) ----
#pragma unroll
        for (int r = 0; r < 8; ++r) {
            float tm = fmaxf(s0[r], s1[r]);
            tm = fmaxf(tm, __shfl_xor(tm, 1, 16));
            tm = fmaxf(tm, __shfl_xor(tm, 2, 16));
            tm = fmaxf(tm, __shfl_xor(tm, 4, 16));
            tm = fmaxf(tm, __shfl_xor(tm, 8, 16));
            float nm   = fmaxf(Mx[r], tm);
            float corr = __expf(Mx[r] - nm);
            float p0 = __expf(s0[r] - nm);
            float p1 = __expf(s1[r] - nm);
            float rs = p0 + p1;
            rs += __shfl_xor(rs, 1, 16);
            rs += __shfl_xor(rs, 2, 16);
            rs += __shfl_xor(rs, 4, 16);
            rs += __shfl_xor(rs, 8, 16);
            Sm[r] = Sm[r] * corr + rs;
            Mx[r] = nm;
            o0[r] *= corr;
            o1[r] *= corr;
            P[(r + hi * 8) * 32 + row]      = p0;    // C-layout -> LDS
            P[(r + hi * 8) * 32 + 16 + row] = p1;
        }
        __syncthreads();
        v16h pa;                                      // re-read as A fragment
#pragma unroll
        for (int e = 0; e < 16; ++e) {
            int k = hi * 16 + e;
            pa[e] = (_Float16)P[row * 32 + k];
        }
        __syncthreads();

        // ---- O += P * V ----------------------------------------------------
        v16h vb0, vb1;
        const float* vp = qkv + ((size_t)(b * SEQ) + ms) * 768 + 512 + h * 32;
#pragma unroll
        for (int e = 0; e < 16; ++e) {
            int kk = hi * 16 + e;                     // key-local index
            vb0[e] = (_Float16)vp[(size_t)kk * 768 + row];
            vb1[e] = (_Float16)vp[(size_t)kk * 768 + 16 + row];
        }
        o0 = __builtin_amdgcn_wmma_f32_16x16x32_f16(false, pa, false, vb0, (short)0, o0, false, false);
        o1 = __builtin_amdgcn_wmma_f32_16x16x32_f16(false, pa, false, vb1, (short)0, o1, false, false);
    }

#pragma unroll
    for (int r = 0; r < 8; ++r) {
        int l = lt * 16 + r + hi * 8;
        float inv = 1.f / Sm[r];
        float* op = o + ((size_t)(b * SEQ) + l) * 256 + h * 32;
        op[row]      = o0[r] * inv;
        op[16 + row] = o1[r] * inv;
    }
}

// ===========================================================================
// Elementwise / norm / scan kernels (f32)
// ===========================================================================
__global__ void f32_to_f16_k(const float* __restrict__ s, _Float16* __restrict__ d, size_t n)
{
    size_t i = (size_t)blockIdx.x * blockDim.x + threadIdx.x;
    size_t st = (size_t)gridDim.x * blockDim.x;
    for (; i < n; i += st) d[i] = (_Float16)s[i];
}

__global__ void gauss_smooth(const float* __restrict__ ts, float* __restrict__ Tsm)
{
    int idx = blockIdx.x * blockDim.x + threadIdx.x;
    if (idx >= BL) return;
    int b = idx / SEQ, l = idx % SEQ;
    float wsum = 0.f;
#pragma unroll
    for (int j = 0; j <= 16; ++j) {
        float dx = (j - 8) * 0.5f;
        wsum += __expf(-0.5f * dx * dx);
    }
    float acc = 0.f;
#pragma unroll
    for (int j = 0; j <= 16; ++j) {
        float dx = (j - 8) * 0.5f;
        float w = __expf(-0.5f * dx * dx) / wsum;
        int m = l + j - 8;
        int mi = (m < 0) ? (-1 - m) : ((m >= SEQ) ? (2 * SEQ - 1 - m) : m);
        acc += w * ts[b * SEQ + mi];
    }
    Tsm[idx] = acc;
}

__global__ void make_pf(const float* __restrict__ Tsm, float* __restrict__ pf)
{
    int idx = blockIdx.x * blockDim.x + threadIdx.x;
    if (idx >= BL) return;
    int b = idx / SEQ, l = idx % SEQ;
    const float* T = Tsm + b * SEQ;
    float tt = T[l] - T[0];
    float td = ((l < SEQ - 1) ? (T[l + 1] - T[l]) : (T[SEQ - 1] - T[SEQ - 2])) / 0.02f;
    float t  = (l == 0) ? 0.02f : (float)l * 0.02f;
    float* p = pf + (size_t)idx * 4;
    p[0] = tt; p[1] = td; p[2] = rsqrtf(t); p[3] = 0.004f;  // 0.02/5.0
}

__global__ void physics_attn(const float* __restrict__ pf,
                             const float* __restrict__ w1, const float* __restrict__ b1,
                             const float* __restrict__ w2, const float* __restrict__ b2,
                             float* __restrict__ pwx)
{
    int idx = blockIdx.x * blockDim.x + threadIdx.x;
    if (idx >= BL) return;
    int b = idx / SEQ, l = idx % SEQ;
    float f[4];
#pragma unroll
    for (int i = 0; i < 4; ++i) f[i] = pf[(size_t)idx * 4 + i];
    float hbuf[64];
    for (int o = 0; o < 64; ++o) {
        float s = b1[o];
#pragma unroll
        for (int i = 0; i < 4; ++i) s += f[i] * w1[o * 4 + i];
        hbuf[o] = fmaxf(s, 0.f);
    }
    float lg[4];
#pragma unroll
    for (int c = 0; c < 4; ++c) {
        float s = b2[c];
        for (int o = 0; o < 64; ++o) s += hbuf[o] * w2[c * 64 + o];
        lg[c] = s;
    }
    float mx = fmaxf(fmaxf(lg[0], lg[1]), fmaxf(lg[2], lg[3]));
    float e0 = __expf(lg[0] - mx), e1 = __expf(lg[1] - mx),
          e2 = __expf(lg[2] - mx), e3 = __expf(lg[3] - mx);
    float inv = 1.f / (e0 + e1 + e2 + e3);
    float pr[4] = { e0 * inv, e1 * inv, e2 * inv, e3 * inv };
#pragma unroll
    for (int c = 0; c < 4; ++c)
        pwx[((size_t)(b * 4 + c)) * SEQ + l] = f[c] * pr[c];
}

__global__ void conv_bn_relu(const float* __restrict__ pwx, const float* __restrict__ w,
                             const float* __restrict__ bias, const float* __restrict__ g,
                             const float* __restrict__ bb, const float* __restrict__ m,
                             const float* __restrict__ v, float* __restrict__ xout,
                             int CO, int KW, int co_off)
{
    int idx = blockIdx.x * blockDim.x + threadIdx.x;
    if (idx >= BSZ * CO * SEQ) return;
    int l = idx % SEQ; int t = idx / SEQ; int c = t % CO; int b = t / CO;
    int pad = (KW - 1) / 2;
    float s = bias[c];
    for (int ci = 0; ci < 4; ++ci)
        for (int j = 0; j < KW; ++j) {
            int ll = l + j - pad;
            if (ll >= 0 && ll < SEQ)
                s += pwx[((size_t)(b * 4 + ci)) * SEQ + ll] * w[(c * 4 + ci) * KW + j];
        }
    float bn = (s - m[c]) * rsqrtf(v[c] + 1e-5f) * g[c] + bb[c];
    xout[((size_t)(b * SEQ + l)) * DM + co_off + c] = fmaxf(bn, 0.f);
}

__global__ void bn_posenc(float* __restrict__ x,
                          const float* __restrict__ g, const float* __restrict__ bb,
                          const float* __restrict__ m, const float* __restrict__ v)
{
    int idx = blockIdx.x * blockDim.x + threadIdx.x;
    if (idx >= BL * DM) return;
    int c = idx & (DM - 1);
    int l = (idx / DM) % SEQ;
    float val = (x[idx] - m[c]) * rsqrtf(v[c] + 1e-5f) * g[c] + bb[c];
    float ang = (float)l * __expf((float)((c >> 1) * 2) * (-9.210340371976184f / 256.f));
    val += (c & 1) ? __cosf(ang) : __sinf(ang);
    x[idx] = val;
}

// out = LN(amul*a + bsum) * g + bb   (D=256, one wave per row, in-place safe)
__global__ __launch_bounds__(256)
void ln_add(const float* __restrict__ a, float amul, const float* __restrict__ bsum,
            const float* __restrict__ g, const float* __restrict__ bb,
            float* __restrict__ out, int rows)
{
    int rowid = blockIdx.x * blockDim.y + threadIdx.y;
    if (rowid >= rows) return;
    int lane = threadIdx.x;
    const float* ar = a + (size_t)rowid * DM;
    const float* br = bsum + (size_t)rowid * DM;
    float vals[8], s = 0.f, s2 = 0.f;
#pragma unroll
    for (int j = 0; j < 8; ++j) {
        float vv = amul * ar[lane + 32 * j] + br[lane + 32 * j];
        vals[j] = vv; s += vv; s2 += vv * vv;
    }
#pragma unroll
    for (int msk = 16; msk >= 1; msk >>= 1) {
        s  += __shfl_xor(s,  msk, 32);
        s2 += __shfl_xor(s2, msk, 32);
    }
    float mean = s * (1.f / 256.f);
    float var  = s2 * (1.f / 256.f) - mean * mean;
    float inv  = rsqrtf(var + 1e-5f);
    float* orow = out + (size_t)rowid * DM;
#pragma unroll
    for (int j = 0; j < 8; ++j) {
        int c = lane + 32 * j;
        orow[c] = (vals[j] - mean) * inv * g[c] + bb[c];
    }
}

__global__ void dwconv_silu(const float* __restrict__ xz, const float* __restrict__ w,
                            const float* __restrict__ bias, float* __restrict__ uc)
{
    int idx = blockIdx.x * blockDim.x + threadIdx.x;
    if (idx >= BL * DI) return;
    int d = idx % DI; int bl = idx / DI; int b = bl / SEQ, l = bl % SEQ;
    float s = bias[d];
#pragma unroll
    for (int j = 0; j < 4; ++j) {
        int ll = l + j - 3;
        if (ll >= 0) s += xz[((size_t)(b * SEQ + ll)) * 1024 + d] * w[d * 4 + j];
    }
    uc[(size_t)idx] = s / (1.f + __expf(-s));
}

__global__ void softplus_bias(float* __restrict__ dt, const float* __restrict__ bias, int n)
{
    int idx = blockIdx.x * blockDim.x + threadIdx.x;
    if (idx >= n) return;
    float v = dt[idx] + bias[idx % DI];
    dt[idx] = (v > 20.f) ? v : log1pf(__expf(v));
}

// Selective scan fused with D-skip + silu(z) gating. y may alias uc.
__global__ void mamba_scan(const float* uc, const float* __restrict__ dtv,
                           const float* __restrict__ dbl, const float* __restrict__ A_log,
                           const float* __restrict__ Dp, const float* __restrict__ xz,
                           float* y)
{
    int idx = blockIdx.x * blockDim.x + threadIdx.x;
    if (idx >= BSZ * DI) return;
    int b = idx / DI, d = idx % DI;
    float A[DS], h[DS];
#pragma unroll
    for (int n = 0; n < DS; ++n) { A[n] = -__expf(A_log[d * DS + n]); h[n] = 0.f; }
    float Dd = Dp[d];
    for (int l = 0; l < SEQ; ++l) {
        size_t r = (size_t)(b * SEQ + l);
        float u  = uc[r * DI + d];
        float dt = dtv[r * DI + d];
        const float* Brow = dbl + r * 80 + DTR;
        const float* Crow = dbl + r * 80 + DTR + DS;
        float acc = 0.f;
#pragma unroll
        for (int n = 0; n < DS; ++n) {
            h[n] = h[n] * __expf(dt * A[n]) + dt * Brow[n] * u;
            acc += h[n] * Crow[n];
        }
        float z = xz[r * 1024 + DI + d];
        y[r * DI + d] = (acc + u * Dd) * (z / (1.f + __expf(-z)));
    }
}

__global__ void bidir_concat(const float* __restrict__ x, float* __restrict__ xb)
{
    int idx = blockIdx.x * blockDim.x + threadIdx.x;
    if (idx >= BL * 512) return;
    int c = idx % 512; int bl = idx / 512; int b = bl / SEQ, l = bl % SEQ;
    xb[idx] = (c < DM) ? x[((size_t)(b * SEQ + l)) * DM + c]
                       : x[((size_t)(b * SEQ + (SEQ - 1 - l))) * DM + (c - DM)];
}

__global__ void pool_mean(const float* __restrict__ x, float* __restrict__ out, int D)
{
    int idx = blockIdx.x * blockDim.x + threadIdx.x;
    if (idx >= BSZ * D) return;
    int b = idx / D, c = idx % D;
    float s = 0.f;
    for (int l = 0; l < SEQ; ++l) s += x[((size_t)(b * SEQ + l)) * D + c];
    out[idx] = s * (1.f / SEQ);
}

template <int ACT>
__global__ void small_linear(const float* __restrict__ A, const float* __restrict__ W,
                             const float* __restrict__ bias, float* __restrict__ out,
                             int M, int N, int K)
{
    int idx = blockIdx.x * blockDim.x + threadIdx.x;
    if (idx >= M * N) return;
    int m = idx / N, n = idx % N;
    float s = bias ? bias[n] : 0.f;
    for (int k = 0; k < K; ++k) s += A[(size_t)m * K + k] * W[(size_t)n * K + k];
    if (ACT == 1) s = fmaxf(s, 0.f);
    out[idx] = s;
}

__global__ void concat3(const float* __restrict__ a, int na, const float* __restrict__ b,
                        int nb, const float* __restrict__ c, int nc,
                        float* __restrict__ out, int M)
{
    int tot = na + nb + nc;
    int idx = blockIdx.x * blockDim.x + threadIdx.x;
    if (idx >= M * tot) return;
    int m = idx / tot, j = idx % tot;
    float v;
    if (j < na) v = a[m * na + j];
    else if (j < na + nb) v = b[m * nb + (j - na)];
    else v = c[m * nc + (j - na - nb)];
    out[idx] = v;
}

__global__ void final_out(const float* __restrict__ fu3, const float* __restrict__ ow,
                          const float* __restrict__ ob, const float* __restrict__ pd3,
                          const float* __restrict__ aw, const float* __restrict__ ab,
                          float* __restrict__ out)
{
    int b = threadIdx.x;
    if (b >= BSZ) return;
    float s = ob[0];
    for (int k = 0; k < 64; ++k) s += fu3[b * 64 + k] * ow[k];
    float s2 = ab[0];
    for (int k = 0; k < 32; ++k) s2 += pd3[b * 32 + k] * aw[k];
    out[b] = s + 0.15f * s2;
}

// ===========================================================================
// Host-side orchestration
// ===========================================================================
extern "C" void kernel_launch(void* const* d_in, const int* in_sizes, int n_in,
                              void* d_out, int out_size, void* d_ws, size_t ws_size,
                              hipStream_t stream)
{
    (void)in_sizes; (void)n_in; (void)out_size; (void)ws_size;
    auto F = [&](int i) -> const float* { return (const float*)d_in[i]; };

    // ---- inputs in setup_inputs() insertion order -------------------------
    int ix = 0;
    const float* ts    = F(ix++);
    const float* pa_w1 = F(ix++); const float* pa_b1 = F(ix++);
    const float* pa_w2 = F(ix++); const float* pa_b2 = F(ix++);
    const float *ms_w[3], *ms_b[3], *bn_g[3], *bn_b[3], *bn_m[3], *bn_v[3];
    for (int i = 0; i < 3; ++i) {
        ms_w[i] = F(ix++); ms_b[i] = F(ix++);
        bn_g[i] = F(ix++); bn_b[i] = F(ix++); bn_m[i] = F(ix++); bn_v[i] = F(ix++);
    }
    const float* inp_w = F(ix++); const float* inp_b = F(ix++);
    const float* ibn_g = F(ix++); const float* ibn_b = F(ix++);
    const float* ibn_m = F(ix++); const float* ibn_v = F(ix++);
    struct MP { const float *in_w,*conv_w,*conv_b,*x_w,*dt_w,*dt_b,*A_log,*D,*out_w,*ng,*nb; } mp[3];
    for (int i = 0; i < 3; ++i) {
        mp[i].in_w=F(ix++); mp[i].conv_w=F(ix++); mp[i].conv_b=F(ix++);
        mp[i].x_w=F(ix++);  mp[i].dt_w=F(ix++);   mp[i].dt_b=F(ix++);
        mp[i].A_log=F(ix++);mp[i].D=F(ix++);      mp[i].out_w=F(ix++);
        mp[i].ng=F(ix++);   mp[i].nb=F(ix++);
    }
    const float* bi_w = F(ix++); const float* bi_b = F(ix++);
    struct TP { const float *qkv_w,*qkv_b,*ow,*ob,*ln1g,*ln1b,*f1w,*f1b,*f2w,*f2b,*ln2g,*ln2b; } tp[3];
    for (int i = 0; i < 3; ++i) {
        tp[i].qkv_w=F(ix++); tp[i].qkv_b=F(ix++); tp[i].ow=F(ix++); tp[i].ob=F(ix++);
        tp[i].ln1g=F(ix++);  tp[i].ln1b=F(ix++);  tp[i].f1w=F(ix++); tp[i].f1b=F(ix++);
        tp[i].f2w=F(ix++);   tp[i].f2b=F(ix++);   tp[i].ln2g=F(ix++); tp[i].ln2b=F(ix++);
    }
    const float* lp_w = F(ix++); const float* lp_b = F(ix++);
    const float* gp_w = F(ix++); const float* gp_b = F(ix++);
    const float* pd1_w = F(ix++); const float* pd1_b = F(ix++);
    const float* pd2_w = F(ix++); const float* pd2_b = F(ix++);
    const float* pd3_w = F(ix++); const float* pd3_b = F(ix++);
    const float* fu1_w = F(ix++); const float* fu1_b = F(ix++);
    const float* fu2_w = F(ix++); const float* fu2_b = F(ix++);
    const float* fu3_w = F(ix++); const float* fu3_b = F(ix++);
    const float* outw = F(ix++); const float* outb = F(ix++);
    const float* auxw = F(ix++); const float* auxb = F(ix++);

    // ---- workspace --------------------------------------------------------
    char* wsp = (char*)d_ws;
    size_t off = 0;
    auto alloc = [&](size_t bytes) -> void* {
        off = (off + 255) & ~(size_t)255;
        void* p = wsp + off;
        off += bytes;
        return p;
    };
    float* Tsm  = (float*)alloc((size_t)BL * 4);
    float* PF   = (float*)alloc((size_t)BL * 4 * 4);
    float* PWX  = (float*)alloc((size_t)BSZ * 4 * SEQ * 4);
    float* XA   = (float*)alloc((size_t)BL * DM * 4);       // feats / residual add
    float* XB   = (float*)alloc((size_t)BL * DM * 4);       // running x
    float* BIG0 = (float*)alloc((size_t)BL * 1024 * 4);     // xz / qkv / ffn-hidden
    float* BIG1 = (float*)alloc((size_t)BL * 512 * 4);      // uc&y / xb / attn-out
    float* DBL  = (float*)alloc((size_t)BL * 80 * 4);
    float* DT   = (float*)alloc((size_t)BL * DI * 4);
    _Float16* HA = (_Float16*)alloc((size_t)BL * 1024 * 2); // f16 A operand
    _Float16* HW = (_Float16*)alloc((size_t)262144 * 2);    // f16 W operand
    float* POOL = (float*)alloc(BSZ * DM * 4);
    float* PS   = (float*)alloc(BSZ * 4 * 4);
    float* XL   = (float*)alloc(BSZ * 128 * 4);
    float* XG   = (float*)alloc(BSZ * 128 * 4);
    float* PD1  = (float*)alloc(BSZ * 128 * 4);
    float* PD2  = (float*)alloc(BSZ * 64 * 4);
    float* PD3  = (float*)alloc(BSZ * 32 * 4);
    float* FU   = (float*)alloc(BSZ * 288 * 4);
    float* FU1  = (float*)alloc(BSZ * 512 * 4);
    float* FU2  = (float*)alloc(BSZ * 256 * 4);
    float* FU3  = (float*)alloc(BSZ * 64 * 4);

    auto blocks = [](long n) { return (unsigned)((n + 255) / 256); };

    auto gemm = [&](const float* Af, int lda, const float* Wf, const float* bias,
                    float* C, int M, int N, int K, int act) {
        size_t nA = (size_t)M * lda;
        size_t nW = (size_t)N * K;
        f32_to_f16_k<<<blocks((long)nA), 256, 0, stream>>>(Af, HA, nA);
        f32_to_f16_k<<<blocks((long)nW), 256, 0, stream>>>(Wf, HW, nW);
        dim3 g(M / 16, (N + 63) / 64), blk(32, 4);
        if (act) gemm_wmma<1><<<g, blk, 0, stream>>>(HA, lda, HW, bias, C, M, N, K);
        else     gemm_wmma<0><<<g, blk, 0, stream>>>(HA, lda, HW, bias, C, M, N, K);
    };

    // ---- stage A: preprocessing ------------------------------------------
    gauss_smooth<<<blocks(BL), 256, 0, stream>>>(ts, Tsm);
    make_pf<<<blocks(BL), 256, 0, stream>>>(Tsm, PF);
    physics_attn<<<blocks(BL), 256, 0, stream>>>(PF, pa_w1, pa_b1, pa_w2, pa_b2, PWX);
    {
        int cos[3] = {64, 64, 128}, ks[3] = {3, 7, 15}, co_off = 0;
        for (int i = 0; i < 3; ++i) {
            long n = (long)BSZ * cos[i] * SEQ;
            conv_bn_relu<<<blocks(n), 256, 0, stream>>>(PWX, ms_w[i], ms_b[i],
                bn_g[i], bn_b[i], bn_m[i], bn_v[i], XA, cos[i], ks[i], co_off);
            co_off += cos[i];
        }
    }
    gemm(XA, DM, inp_w, inp_b, XB, BL, DM, DM, 0);
    bn_posenc<<<blocks((long)BL * DM), 256, 0, stream>>>(XB, ibn_g, ibn_b, ibn_m, ibn_v);

    // ---- stage B: Mamba x3  (note: xres == x, so LN(2x + mamba(x))) ------
    for (int i = 0; i < 3; ++i) {
        gemm(XB, DM, mp[i].in_w, nullptr, BIG0, BL, 1024, DM, 0);          // xz
        dwconv_silu<<<blocks((long)BL * DI), 256, 0, stream>>>(BIG0, mp[i].conv_w, mp[i].conv_b, BIG1);
        gemm(BIG1, DI, mp[i].x_w, nullptr, DBL, BL, 80, DI, 0);            // dtr|B|C
        gemm(DBL, 80, mp[i].dt_w, nullptr, DT, BL, DI, DTR, 0);            // dt pre-act
        softplus_bias<<<blocks((long)BL * DI), 256, 0, stream>>>(DT, mp[i].dt_b, BL * DI);
        mamba_scan<<<blocks(BSZ * DI), 256, 0, stream>>>(BIG1, DT, DBL, mp[i].A_log, mp[i].D, BIG0, BIG1);
        gemm(BIG1, DI, mp[i].out_w, nullptr, XA, BL, DM, DI, 0);           // out proj
        ln_add<<<BL / 8, dim3(32, 8), 0, stream>>>(XB, 2.f, XA, mp[i].ng, mp[i].nb, XB, BL);
    }

    // ---- stage C: bidirectional projection -------------------------------
    bidir_concat<<<blocks((long)BL * 512), 256, 0, stream>>>(XB, BIG1);
    gemm(BIG1, 512, bi_w, bi_b, XB, BL, DM, 512, 0);

    // ---- stage D: transformer x3 -----------------------------------------
    for (int i = 0; i < 3; ++i) {
        gemm(XB, DM, tp[i].qkv_w, tp[i].qkv_b, BIG0, BL, 768, DM, 0);
        attention_flash<<<dim3(SEQ / 16, 8, BSZ), 32, 0, stream>>>(BIG0, BIG1);
        gemm(BIG1, DM, tp[i].ow, tp[i].ob, XA, BL, DM, DM, 0);
        ln_add<<<BL / 8, dim3(32, 8), 0, stream>>>(XB, 1.f, XA, tp[i].ln1g, tp[i].ln1b, XB, BL);
        gemm(XB, DM, tp[i].f1w, tp[i].f1b, BIG0, BL, 1024, DM, 1);
        gemm(BIG0, 1024, tp[i].f2w, tp[i].f2b, XA, BL, DM, 1024, 0);
        ln_add<<<BL / 8, dim3(32, 8), 0, stream>>>(XB, 1.f, XA, tp[i].ln2g, tp[i].ln2b, XB, BL);
    }

    // ---- stage E: heads ---------------------------------------------------
    pool_mean<<<blocks(BSZ * DM), 256, 0, stream>>>(XB, POOL, DM);
    pool_mean<<<1, 64, 0, stream>>>(PF, PS, 4);
    small_linear<0><<<blocks(BSZ * 128), 256, 0, stream>>>(POOL, lp_w, lp_b, XL, BSZ, 128, DM);
    small_linear<0><<<blocks(BSZ * 128), 256, 0, stream>>>(POOL, gp_w, gp_b, XG, BSZ, 128, DM);
    small_linear<1><<<blocks(BSZ * 128), 256, 0, stream>>>(PS,  pd1_w, pd1_b, PD1, BSZ, 128, 4);
    small_linear<1><<<blocks(BSZ * 64),  256, 0, stream>>>(PD1, pd2_w, pd2_b, PD2, BSZ, 64, 128);
    small_linear<0><<<blocks(BSZ * 32),  256, 0, stream>>>(PD2, pd3_w, pd3_b, PD3, BSZ, 32, 64);
    concat3<<<blocks(BSZ * 288), 256, 0, stream>>>(XL, 128, XG, 128, PD3, 32, FU, BSZ);
    small_linear<1><<<blocks(BSZ * 512), 256, 0, stream>>>(FU,  fu1_w, fu1_b, FU1, BSZ, 512, 288);
    small_linear<1><<<blocks(BSZ * 256), 256, 0, stream>>>(FU1, fu2_w, fu2_b, FU2, BSZ, 256, 512);
    small_linear<1><<<blocks(BSZ * 64),  256, 0, stream>>>(FU2, fu3_w, fu3_b, FU3, BSZ, 64, 256);
    final_out<<<1, 32, 0, stream>>>(FU3, outw, outb, PD3, auxw, auxb, (float*)d_out);
}
